// Attention_86646670230179
// MI455X (gfx1250) — matrix operand order repaired and checked by reading, not compiled, not run on hardware
//
#include <hip/hip_runtime.h>

typedef __attribute__((ext_vector_type(4)))  unsigned int v4u;
typedef __attribute__((ext_vector_type(4)))  int          v4i;
typedef __attribute__((ext_vector_type(8)))  int          v8i;
typedef __attribute__((ext_vector_type(4)))  float        v4f;
typedef __attribute__((ext_vector_type(8)))  float        v8f;
typedef __attribute__((ext_vector_type(8)))  __bf16       v8bf;
typedef __attribute__((ext_vector_type(16))) __bf16       v16bf;

union FragU {
  v4u   u[2];
  v16bf v;
};

static __device__ inline v8f v8zero() {
  v8f z = {0.f, 0.f, 0.f, 0.f, 0.f, 0.f, 0.f, 0.f};
  return z;
}

// Problem dims (compile-time so index math folds to shifts/masks)
constexpr int cB = 2, cL = 2048, cS = 2048, cE = 1024, cH = 16, cD = 64;
constexpr int cM = cB * cL;   // 4096

#define TM 128
#define TN 128
#define TK 32

// 32-bit LDS offset of a generic pointer to __shared__ memory
// (LDS aperture: addr[31:0] is the LDS byte address per ISA 10.2)
static __device__ inline unsigned lds_off32(const void* p) {
  return (unsigned)(unsigned long long)p;
}

static __device__ inline void async_copy_b128(unsigned lds_byte_addr, const void* gptr) {
  asm volatile("global_load_async_to_lds_b128 %0, %1, off"
               :: "v"(lds_byte_addr), "v"((unsigned long long)gptr)
               : "memory");
}

static __device__ inline void wait_async0() {
  asm volatile("s_wait_asynccnt 0x0" ::: "memory");
}

// One 2D TDM descriptor: copy tileRows x tileCols (bf16, row stride
// rowStrideElems) from global to contiguous LDS at ldsAddr.
// D# layout per CDNA5 ISA 8.3/8.4 (group0: count|lds|global|type,
// group1: data_size=2B, tensor dims, tile dims, dim0 stride).
// This toolchain has the 6-arg builtin: (g0, g1, g2, g3, g4, cpol).
static __device__ inline void tdm_load_2d(unsigned ldsAddr, const void* gptr,
                                          unsigned tileRows, unsigned tileCols,
                                          unsigned rowStrideElems,
                                          unsigned tensorD0, unsigned tensorD1) {
  unsigned long long ga = (unsigned long long)gptr;
  v4u g0;
  g0[0] = 1u;                                               // count=1 (user D#)
  g0[1] = ldsAddr;                                          // lds_addr [63:32]
  g0[2] = (unsigned)ga;                                     // global_addr lo
  g0[3] = (unsigned)((ga >> 32) & 0x01FFFFFFu) | (2u << 30);// addr hi | type=2
  v8i g1;
  g1[0] = (int)(1u << 16);                                  // data_size=1 -> 2B
  g1[1] = (int)((tensorD0 & 0xFFFFu) << 16);                // tensor_dim0 lo
  g1[2] = (int)(((tensorD0 >> 16) & 0xFFFFu) |
                ((tensorD1 & 0xFFFFu) << 16));              // td0 hi | td1 lo
  g1[3] = (int)(((tensorD1 >> 16) & 0xFFFFu) |
                ((tileCols & 0xFFFFu) << 16));              // td1 hi | tile_dim0
  g1[4] = (int)(tileRows & 0xFFFFu);                        // tile_dim1 (td2=0)
  g1[5] = (int)rowStrideElems;                              // dim0_stride lo
  g1[6] = 0;                                                // stride hi / d1s
  g1[7] = 0;
  v4i z4 = {0, 0, 0, 0};
  v8i z8 = {0, 0, 0, 0, 0, 0, 0, 0};
  __builtin_amdgcn_tensor_load_to_lds(g0, g1, z4, z4, z8, 0);
}

static __device__ inline void wait_tensor0() {
  __builtin_amdgcn_s_wait_tensorcnt((short)0);
}

// ---------------------------------------------------------------------------
// Elementwise f32 -> bf16 convert (8 elements/thread, b128 in / b128 out)
// ---------------------------------------------------------------------------
__global__ __launch_bounds__(256) void cvt_bf16(const float* __restrict__ x,
                                                __bf16* __restrict__ y) {
  const size_t i = ((size_t)blockIdx.x * 256 + threadIdx.x) * 8;
  v4f a = *(const v4f*)(x + i);
  v4f b = *(const v4f*)(x + i + 4);
  v8bf o;
  o[0] = (__bf16)a.x; o[1] = (__bf16)a.y; o[2] = (__bf16)a.z; o[3] = (__bf16)a.w;
  o[4] = (__bf16)b.x; o[5] = (__bf16)b.y; o[6] = (__bf16)b.z; o[7] = (__bf16)b.w;
  *(v8bf*)(y + i) = o;
}

// ---------------------------------------------------------------------------
// Weight transpose + convert: Wt[n][k] = (bf16) W[k][n], E x E
// ---------------------------------------------------------------------------
__global__ __launch_bounds__(256) void cvt_transpose(const float* __restrict__ W,
                                                     __bf16* __restrict__ Wt) {
  __shared__ float T[32][33];
  const int k0 = blockIdx.x * 32, n0 = blockIdx.y * 32;
  const int tid = threadIdx.x;
  const int r = tid >> 3;          // 0..31
  const int c = (tid & 7) * 4;     // 0..28
  v4f d = *(const v4f*)(W + (size_t)(k0 + r) * cE + n0 + c);
  T[r][c + 0] = d.x; T[r][c + 1] = d.y; T[r][c + 2] = d.z; T[r][c + 3] = d.w;
  __syncthreads();
  __bf16* out = Wt + (size_t)(n0 + r) * cE + k0 + c;
  out[0] = (__bf16)T[c + 0][r];
  out[1] = (__bf16)T[c + 1][r];
  out[2] = (__bf16)T[c + 2][r];
  out[3] = (__bf16)T[c + 3][r];
}

// ---------------------------------------------------------------------------
// C[M,N] = X[M,K] @ Wt[N,K]^T + bias  (bf16 WMMA, f32 accum)
// A tile staged by TDM (TENSORcnt), B tile by per-lane async (ASYNCcnt);
// double-buffered LDS.
// MODE 0: f32 [M,N]; MODE 1: bf16 [B,H,L,D]; MODE 2: bf16 [B,H,D,S]
// ---------------------------------------------------------------------------
template <int MODE>
__global__ __launch_bounds__(256) void wmma_gemm(
    const __bf16* __restrict__ X, const __bf16* __restrict__ Wt,
    const float* __restrict__ bias, void* __restrict__ Out) {
  constexpr int N = cE, K = cE;

  __shared__ __bf16 As[2][TM][TK];   // 2 x 8 KB
  __shared__ __bf16 Bs[2][TN][TK];   // 2 x 8 KB

  const int tid  = threadIdx.x;
  const int wv   = tid >> 5;
  const int lane = tid & 31;
  const int hlf  = lane >> 4;
  const int l16  = lane & 15;
  const int wm   = wv & 3;
  const int wn   = wv >> 2;
  const int m0   = blockIdx.x * TM;
  const int n0   = blockIdx.y * TN;

  v8f acc[2][4];
#pragma unroll
  for (int i = 0; i < 2; ++i)
#pragma unroll
    for (int j = 0; j < 4; ++j) acc[i][j] = v8zero();

  auto stage = [&](int buf, int k0) {
    // B tile: per-lane async global->LDS b128 copies
#pragma unroll
    for (int it = 0; it < 2; ++it) {
      int i = it * 256 + tid;        // 0..511 b128 chunks
      int r = i >> 2;                // 0..127
      int c = (i & 3) * 8;           // bf16 elems
      async_copy_b128(lds_off32(&Bs[buf][r][c]),
                      Wt + (size_t)(n0 + r) * K + k0 + c);
    }
    // A tile: single TDM 2D descriptor, issued by wave 0
    if (wv == 0) {
      tdm_load_2d(lds_off32(&As[buf][0][0]),
                  X + (size_t)m0 * K + k0,
                  TM, TK, K, (unsigned)K, (unsigned)TM);
    }
  };

  stage(0, 0);
  wait_async0();
  if (wv == 0) wait_tensor0();
  __syncthreads();

  int buf = 0;
  for (int k0 = 0; k0 < K; k0 += TK) {
    if (k0 + TK < K) stage(buf ^ 1, k0 + TK);   // prefetch next slab

    FragU a[2];
#pragma unroll
    for (int mt = 0; mt < 2; ++mt) {
      int row = wm * 32 + mt * 16 + l16;
      a[mt].u[0] = *(const v4u*)(&As[buf][row][hlf * 8]);
      a[mt].u[1] = *(const v4u*)(&As[buf][row][16 + hlf * 8]);
    }
#pragma unroll
    for (int nt = 0; nt < 4; ++nt) {
      int col = wn * 64 + nt * 16 + l16;
      FragU b;
      b.u[0] = *(const v4u*)(&Bs[buf][col][hlf * 8]);
      b.u[1] = *(const v4u*)(&Bs[buf][col][16 + hlf * 8]);
#pragma unroll
      for (int mt = 0; mt < 2; ++mt)
        acc[mt][nt] = __builtin_amdgcn_wmma_f32_16x16x32_bf16(
            false, a[mt].v, false, b.v, (short)0, acc[mt][nt], false, false);
    }

    wait_async0();                     // our B-tile asyncs done
    if (wv == 0) wait_tensor0();       // A-tile TDM done
    __syncthreads();
    buf ^= 1;
  }

  // epilogue: bias + per-mode store (all divisors compile-time pow2)
#pragma unroll
  for (int mt = 0; mt < 2; ++mt) {
#pragma unroll
    for (int nt = 0; nt < 4; ++nt) {
#pragma unroll
      for (int j = 0; j < 8; ++j) {
        int gm = m0 + wm * 32 + mt * 16 + j + hlf * 8;
        int gn = n0 + wn * 64 + nt * 16 + l16;
        float v = acc[mt][nt][j] + bias[gn];
        if constexpr (MODE == 0) {
          ((float*)Out)[(size_t)gm * N + gn] = v;
        } else if constexpr (MODE == 1) {
          int b = gm / cL, l = gm % cL;
          int h = gn / cD, d = gn % cD;
          ((__bf16*)Out)[(((size_t)b * cH + h) * cL + l) * cD + d] = (__bf16)v;
        } else {
          int b = gm / cS, s = gm % cS;
          int h = gn / cD, d = gn % cD;
          ((__bf16*)Out)[(((size_t)b * cH + h) * cD + d) * cS + s] = (__bf16)v;
        }
      }
    }
  }
}

// ---------------------------------------------------------------------------
// Flash attention. qh,kh: [B,H,N,D] bf16; vt: [B,H,D,S] bf16; o: [B,L,H,D].
// One block = 64 L-rows of one (b,h); 4 waves x 16 rows.
// ---------------------------------------------------------------------------
__global__ __launch_bounds__(128) void wmma_attn(
    const __bf16* __restrict__ qh, const __bf16* __restrict__ kh,
    const __bf16* __restrict__ vt, __bf16* __restrict__ o) {
  __shared__ __bf16 Ps[4][16][64];

  const int tid  = threadIdx.x;
  const int wv   = tid >> 5;
  const int lane = tid & 31;
  const int hlf  = lane >> 4;
  const int l16  = lane & 15;
  const int bh   = blockIdx.y;
  const int l0   = blockIdx.x * 64 + wv * 16;

  const __bf16* Qb = qh + ((size_t)bh * cL + l0) * cD;
  const __bf16* Kb = kh + (size_t)bh * cS * cD;
  const __bf16* Vb = vt + (size_t)bh * cD * cS;

  FragU qf[2];
#pragma unroll
  for (int kc = 0; kc < 2; ++kc) {
    const __bf16* p = Qb + (size_t)l16 * cD + kc * 32 + hlf * 8;
    qf[kc].u[0] = *(const v4u*)(p);
    qf[kc].u[1] = *(const v4u*)(p + 16);
  }

  float mrun[8], lrun[8];
  v8f oacc[4];
#pragma unroll
  for (int j = 0; j < 8; ++j) { mrun[j] = -1e30f; lrun[j] = 0.f; }
#pragma unroll
  for (int dt = 0; dt < 4; ++dt) oacc[dt] = v8zero();

  const float SCL = 0.125f;   // 1/sqrt(64)
  const float L2E = 1.44269504f;

  for (int s0 = 0; s0 < cS; s0 += 64) {
    if (s0 + 64 < cS) {   // prefetch next chunk of K / V rows
      __builtin_prefetch(Kb + (size_t)(s0 + 64) * cD, 0, 1);
      __builtin_prefetch(Vb + (size_t)l16 * cS + s0 + 64, 0, 1);
    }

    // ---- preload all 8 K fragments (one big load clause) ----
    FragU kf[4][2];
#pragma unroll
    for (int nt = 0; nt < 4; ++nt)
#pragma unroll
      for (int kc = 0; kc < 2; ++kc) {
        const __bf16* p =
            Kb + (size_t)(s0 + nt * 16 + l16) * cD + kc * 32 + hlf * 8;
        kf[nt][kc].u[0] = *(const v4u*)(p);
        kf[nt][kc].u[1] = *(const v4u*)(p + 16);
      }

    // ---- scores ----
    v8f sacc[4];
#pragma unroll
    for (int nt = 0; nt < 4; ++nt) sacc[nt] = v8zero();
#pragma unroll
    for (int nt = 0; nt < 4; ++nt)
#pragma unroll
      for (int kc = 0; kc < 2; ++kc)
        sacc[nt] = __builtin_amdgcn_wmma_f32_16x16x32_bf16(
            false, qf[kc].v, false, kf[nt][kc].v, (short)0, sacc[nt],
            false, false);

    // ---- preload V fragments; they fly during the softmax VALU work ----
    FragU vf[2][4];
#pragma unroll
    for (int sc = 0; sc < 2; ++sc)
#pragma unroll
      for (int dt = 0; dt < 4; ++dt) {
        const __bf16* vp =
            Vb + (size_t)(dt * 16 + l16) * cS + s0 + sc * 32 + hlf * 8;
        vf[sc][dt].u[0] = *(const v4u*)(vp);
        vf[sc][dt].u[1] = *(const v4u*)(vp + 16);
      }

    // ---- online softmax (rows map to VGPR j + lane-half) ----
    float fac[8];
#pragma unroll
    for (int j = 0; j < 8; ++j) {
      float cm = -1e30f;
#pragma unroll
      for (int nt = 0; nt < 4; ++nt) {
        sacc[nt][j] *= SCL;
        cm = fmaxf(cm, sacc[nt][j]);
      }
#pragma unroll
      for (int msk = 1; msk < 16; msk <<= 1)
        cm = fmaxf(cm, __shfl_xor(cm, msk, 32));
      float mnew = fmaxf(mrun[j], cm);
      fac[j] = exp2f((mrun[j] - mnew) * L2E);
      float part = 0.f;
#pragma unroll
      for (int nt = 0; nt < 4; ++nt) {
        float pv = exp2f((sacc[nt][j] - mnew) * L2E);
        part += pv;
        Ps[wv][j + hlf * 8][nt * 16 + l16] = (__bf16)pv;
      }
#pragma unroll
      for (int msk = 1; msk < 16; msk <<= 1)
        part += __shfl_xor(part, msk, 32);
      lrun[j] = lrun[j] * fac[j] + part;
      mrun[j] = mnew;
    }
#pragma unroll
    for (int dt = 0; dt < 4; ++dt)
#pragma unroll
      for (int j = 0; j < 8; ++j) oacc[dt][j] *= fac[j];

    // ---- PV: P from LDS in A-layout, V already in registers ----
#pragma unroll
    for (int sc = 0; sc < 2; ++sc) {
      FragU pf;
      const __bf16* pp = &Ps[wv][l16][sc * 32 + hlf * 8];
      pf.u[0] = *(const v4u*)(pp);
      pf.u[1] = *(const v4u*)(pp + 16);
#pragma unroll
      for (int dt = 0; dt < 4; ++dt)
        oacc[dt] = __builtin_amdgcn_wmma_f32_16x16x32_bf16(
            false, pf.v, false, vf[sc][dt].v, (short)0, oacc[dt],
            false, false);
    }
  }

  const int b = bh / cH, h = bh % cH;
#pragma unroll
  for (int dt = 0; dt < 4; ++dt)
#pragma unroll
    for (int j = 0; j < 8; ++j) {
      int row = l0 + j + hlf * 8;
      int d   = dt * 16 + l16;
      float v = oacc[dt][j] / lrun[j];
      o[(((size_t)b * cL + row) * cH + h) * cD + d] = (__bf16)v;
    }
}

extern "C" void kernel_launch(void* const* d_in, const int* in_sizes, int n_in,
                              void* d_out, int out_size, void* d_ws, size_t ws_size,
                              hipStream_t stream) {
  (void)in_sizes; (void)n_in; (void)out_size; (void)ws_size;

  const float* q  = (const float*)d_in[0];
  const float* k  = (const float*)d_in[1];
  const float* v  = (const float*)d_in[2];
  const float* Wq = (const float*)d_in[3];
  const float* bq = (const float*)d_in[4];
  const float* Wk = (const float*)d_in[5];
  const float* bk = (const float*)d_in[6];
  const float* Wv = (const float*)d_in[7];
  const float* bv = (const float*)d_in[8];
  const float* Wo = (const float*)d_in[9];
  const float* bo = (const float*)d_in[10];

  char* ws = (char*)d_ws;
  const size_t szX = (size_t)cM * cE * sizeof(__bf16);   // 8 MB
  const size_t szW = (size_t)cE * cE * sizeof(__bf16);   // 2 MB
  __bf16* qx  = (__bf16*)(ws);
  __bf16* kx  = (__bf16*)(ws + szX);
  __bf16* vx  = (__bf16*)(ws + 2 * szX);
  __bf16* Wqt = (__bf16*)(ws + 3 * szX);
  __bf16* Wkt = (__bf16*)(ws + 3 * szX + szW);
  __bf16* Wvt = (__bf16*)(ws + 3 * szX + 2 * szW);
  __bf16* Wot = (__bf16*)(ws + 3 * szX + 3 * szW);
  __bf16* qh  = (__bf16*)(ws + 3 * szX + 4 * szW);
  __bf16* khb = (__bf16*)(ws + 4 * szX + 4 * szW);
  __bf16* vtb = (__bf16*)(ws + 5 * szX + 4 * szW);
  __bf16* ob  = (__bf16*)(ws + 6 * szX + 4 * szW);

  const int nX = cM * cE;                       // 4M elements
  const int cvtBlocks = nX / (256 * 8);         // 2048
  cvt_bf16<<<cvtBlocks, 256, 0, stream>>>(q, qx);
  cvt_bf16<<<cvtBlocks, 256, 0, stream>>>(k, kx);
  cvt_bf16<<<cvtBlocks, 256, 0, stream>>>(v, vx);

  dim3 gt(cE / 32, cE / 32);
  cvt_transpose<<<gt, 256, 0, stream>>>(Wq, Wqt);
  cvt_transpose<<<gt, 256, 0, stream>>>(Wk, Wkt);
  cvt_transpose<<<gt, 256, 0, stream>>>(Wv, Wvt);
  cvt_transpose<<<gt, 256, 0, stream>>>(Wo, Wot);

  dim3 gg(cM / TM, cE / TN);
  wmma_gemm<1><<<gg, 256, 0, stream>>>(qx, Wqt, bq, qh);
  wmma_gemm<1><<<gg, 256, 0, stream>>>(kx, Wkt, bk, khb);
  wmma_gemm<2><<<gg, 256, 0, stream>>>(vx, Wvt, bv, vtb);

  dim3 ga(cL / 64, cB * cH);
  wmma_attn<<<ga, 128, 0, stream>>>(qh, khb, vtb, ob);

  wmma_gemm<0><<<gg, 256, 0, stream>>>(ob, Wot, bo, d_out);
}
